// PolynomialTensorRegression_21088289423539
// MI455X (gfx1250) — matrix operand order repaired and checked by reading, not compiled, not run on hardware
//
#include <hip/hip_runtime.h>
#include <math.h>

// ---------------------------------------------------------------------------
// PolynomialTensorRegression forward, fused, bf16 WMMA (gfx1250 / MI455X)
//   preds[b] = beta + C0*sum_r d0 + C1*sum_r d1*d2 + C2*sum_r d3*d4*d5
//   d_k = X @ U_k   (six 32768x1024 * 1024x128 GEMMs sharing X)
//   reg = 0.01 * (|C0|+|C1|+|C2|)
// Software-pipelined K loop: prefetch next K-step's LDS fragments while the
// current step's 12 WMMAs issue.
// ---------------------------------------------------------------------------

typedef __attribute__((ext_vector_type(16))) __bf16 v16bf;
typedef __attribute__((ext_vector_type(8)))  __bf16 v8bf;
typedef __attribute__((ext_vector_type(8)))  float  v8f;

#define B_ROWS   32768
#define D_DIM    1024
#define R_DIM    128
#define BLK_ROWS 128                       // rows per workgroup (8 waves: 4 M x 2 N)
#define KC       128                       // K chunk staged per barrier
#define X_STRIDE 1032                      // bf16 elems per X row in LDS (1024 + 8 pad)
#define U_STRIDE 144                       // bf16 elems per U row in LDS (128 + 16 pad)
#define XLDS_BYTES (BLK_ROWS * X_STRIDE * 2)   // 264192
#define ULDS_BYTES (6 * 32 * U_STRIDE * 2)     // 55296
#define SMEM_BYTES (XLDS_BYTES + ULDS_BYTES)   // 319488 (<= 320KB/WG per CDNA5 ISA)

#define CONCAT16(lo, hi) \
  __builtin_shufflevector(lo, hi, 0,1,2,3,4,5,6,7,8,9,10,11,12,13,14,15)

__device__ __forceinline__ unsigned pk_bf16(float a, float b) {
  unsigned ua = __builtin_bit_cast(unsigned, a);
  unsigned ub = __builtin_bit_cast(unsigned, b);
  unsigned ra = (ua + 0x7FFFu + ((ua >> 16) & 1u)) >> 16;   // RNE
  unsigned rb = (ub + 0x7FFFu + ((ub >> 16) & 1u)) >> 16;
  return ra | (rb << 16);
}

// Stage KC x 32 columns of one U matrix (fp32 global -> bf16, transposed) into
// ulds[(j*32 + nLocal)*U_STRIDE + kk].  nLocal 0..15 -> cols t*16+n (N-half 0),
// nLocal 16..31 -> cols 64+t*16+n (N-half 1).
__device__ __forceinline__ void fill_u(__bf16* __restrict__ ulds,
                                       const float* __restrict__ U,
                                       int j, int t, int kc, int tid) {
  #pragma unroll
  for (int it = 0; it < 8; ++it) {
    int idx = tid + it * 256;                 // 0..2047 bf16x2 pairs
    int kp  = idx >> 5;                       // 0..63  (pair of consecutive k)
    int nl  = idx & 31;                       // 0..31
    int col = (nl < 16) ? (t * 16 + nl) : (64 + t * 16 + (nl - 16));
    int k0  = kc + kp * 2;
    float g0 = U[(size_t)k0 * R_DIM + col];
    float g1 = U[(size_t)(k0 + 1) * R_DIM + col];
    ((unsigned*)(ulds + (j * 32 + nl) * U_STRIDE))[kp] = pk_bf16(g0, g1);
  }
}

// Load one K-step's fragment set: 2 A fragments (two 16-row subtiles) and
// 6 B fragments (one per U matrix).  16x ds_load_b128 total.
__device__ __forceinline__ void load_frags(const __bf16* __restrict__ xlds,
                                           const __bf16* __restrict__ ulds,
                                           int mq, int nq, int l16, int hlf,
                                           int kg, int ks,
                                           v16bf (&af)[2], v16bf (&bf)[6]) {
  #pragma unroll
  for (int a = 0; a < 2; ++a) {
    // 16-bit A layout: lane row = l16, K chunks at (kg+hlf*8) and +16
    const __bf16* ap = xlds + (mq * 32 + a * 16 + l16) * X_STRIDE + kg + hlf * 8;
    v8bf lo = *(const v8bf*)(ap);
    v8bf hi = *(const v8bf*)(ap + 16);
    af[a] = CONCAT16(lo, hi);
  }
  #pragma unroll
  for (int j = 0; j < 6; ++j) {
    // 16-bit B layout: lane col = l16, K chunks at (ks+hlf*8) and +16
    const __bf16* bp = ulds + (j * 32 + nq * 16 + l16) * U_STRIDE + ks + hlf * 8;
    v8bf lo = *(const v8bf*)(bp);
    v8bf hi = *(const v8bf*)(bp + 16);
    bf[j] = CONCAT16(lo, hi);
  }
}

__global__ __launch_bounds__(256, 1)
void poly_reg_fused(const float* __restrict__ X,
                    const float* __restrict__ U00, const float* __restrict__ U10,
                    const float* __restrict__ U11, const float* __restrict__ U20,
                    const float* __restrict__ U21, const float* __restrict__ U22,
                    const float* __restrict__ C,  const float* __restrict__ beta,
                    float* __restrict__ out) {
  extern __shared__ char smem[];
  __bf16* xlds = (__bf16*)smem;
  __bf16* ulds = (__bf16*)(smem + XLDS_BYTES);

  const int tid  = threadIdx.x;
  const int lane = tid & 31;
  const int wave = tid >> 5;
  const int hlf  = lane >> 4;      // lane half (0/1)
  const int l16  = lane & 15;
  const int mq   = wave & 3;       // M quadrant: rows mq*32 .. +31
  const int nq   = wave >> 2;      // N half: r-cols [nq*64, nq*64+64)
  const int rowBase = blockIdx.x * BLK_ROWS;

  // ---- Stage X tile: 128 x 1024 fp32 -> bf16 LDS (read X from HBM once) ----
  {
    const float4* xg = (const float4*)(X + (size_t)rowBase * D_DIM);
    #pragma unroll 4
    for (int it = 0; it < 128; ++it) {
      int q   = tid + it * 256;          // float4 index, 0..32767
      int row = q >> 8;                  // 256 float4 per row
      int c4  = q & 255;
      float4 v = xg[q];
      unsigned* dst = (unsigned*)(xlds + row * X_STRIDE + c4 * 4);
      dst[0] = pk_bf16(v.x, v.y);
      dst[1] = pk_bf16(v.z, v.w);
    }
  }

  const float c0 = C[0], c1 = C[1], c2 = C[2];

  float rp[2][8];                       // per-row partial sums (2 M-subtiles)
  #pragma unroll
  for (int a = 0; a < 2; ++a)
    #pragma unroll
    for (int i = 0; i < 8; ++i) rp[a][i] = 0.0f;

  const v8f zacc = {0.f,0.f,0.f,0.f,0.f,0.f,0.f,0.f};

  #pragma unroll 1
  for (int t = 0; t < 4; ++t) {         // 4 N-tiles of 16 cols per N-half
    v8f acc[2][6];
    #pragma unroll
    for (int a = 0; a < 2; ++a)
      #pragma unroll
      for (int j = 0; j < 6; ++j) acc[a][j] = zacc;

    #pragma unroll 1
    for (int kc = 0; kc < D_DIM; kc += KC) {
      __syncthreads();                  // prior compute done before overwrite
      fill_u(ulds, U00, 0, t, kc, tid);
      fill_u(ulds, U10, 1, t, kc, tid);
      fill_u(ulds, U11, 2, t, kc, tid);
      fill_u(ulds, U20, 3, t, kc, tid);
      fill_u(ulds, U21, 4, t, kc, tid);
      fill_u(ulds, U22, 5, t, kc, tid);
      __syncthreads();

      // ---- software-pipelined K-steps: prefetch next frags during WMMAs ----
      v16bf afC[2], bfC[6];
      load_frags(xlds, ulds, mq, nq, l16, hlf, kc + 0, 0, afC, bfC);
      #pragma unroll
      for (int s = 0; s < 3; ++s) {
        v16bf afN[2], bfN[6];
        load_frags(xlds, ulds, mq, nq, l16, hlf,
                   kc + (s + 1) * 32, (s + 1) * 32, afN, bfN);
        #pragma unroll
        for (int j = 0; j < 6; ++j)
          #pragma unroll
          for (int a = 0; a < 2; ++a)
            acc[a][j] = __builtin_amdgcn_wmma_f32_16x16x32_bf16(
                false, afC[a], false, bfC[j], (short)0, acc[a][j], false, false);
        #pragma unroll
        for (int a = 0; a < 2; ++a) afC[a] = afN[a];
        #pragma unroll
        for (int j = 0; j < 6; ++j) bfC[j] = bfN[j];
      }
      #pragma unroll
      for (int j = 0; j < 6; ++j)
        #pragma unroll
        for (int a = 0; a < 2; ++a)
          acc[a][j] = __builtin_amdgcn_wmma_f32_16x16x32_bf16(
              false, afC[a], false, bfC[j], (short)0, acc[a][j], false, false);
    }

    // Combine the 6 dot-product tiles; fold 16 r-columns into per-row partials.
    // C/D layout: VGPR i, lane -> row = i + 8*hlf (within 16-row subtile).
    #pragma unroll
    for (int a = 0; a < 2; ++a)
      #pragma unroll
      for (int i = 0; i < 8; ++i)
        rp[a][i] += c0 * acc[a][0][i]
                  + c1 * acc[a][1][i] * acc[a][2][i]
                  + c2 * acc[a][3][i] * acc[a][4][i] * acc[a][5][i];
  }

  // ---- Reduce over the 16 lanes of each half (sum over r-columns) ----
  #pragma unroll
  for (int a = 0; a < 2; ++a)
    #pragma unroll
    for (int i = 0; i < 8; ++i) {
      float v = rp[a][i];
      v += __shfl_xor(v, 1);
      v += __shfl_xor(v, 2);
      v += __shfl_xor(v, 4);
      v += __shfl_xor(v, 8);
      rp[a][i] = v;
    }

  // ---- Cross-N-half reduction via LDS (waves nq=0 stash, nq=1 finish) ----
  __syncthreads();                       // LDS no longer needed for tiles
  float* red = (float*)smem;             // 128 floats, one per block row
  if (nq == 0 && l16 == 0) {
    #pragma unroll
    for (int a = 0; a < 2; ++a)
      #pragma unroll
      for (int i = 0; i < 8; ++i)
        red[mq * 32 + a * 16 + hlf * 8 + i] = rp[a][i];
  }
  __syncthreads();
  const float b0 = beta[0];
  if (nq == 1 && l16 == 0) {
    #pragma unroll
    for (int a = 0; a < 2; ++a)
      #pragma unroll
      for (int i = 0; i < 8; ++i) {
        int r = mq * 32 + a * 16 + hlf * 8 + i;
        out[rowBase + r] = b0 + rp[a][i] + red[r];
      }
  }

  if (blockIdx.x == 0 && tid == 0)
    out[B_ROWS] = 0.01f * (fabsf(c0) + fabsf(c1) + fabsf(c2));
}

extern "C" void kernel_launch(void* const* d_in, const int* in_sizes, int n_in,
                              void* d_out, int out_size, void* d_ws, size_t ws_size,
                              hipStream_t stream) {
  (void)in_sizes; (void)n_in; (void)out_size; (void)d_ws; (void)ws_size;
  const float* X   = (const float*)d_in[0];
  const float* U00 = (const float*)d_in[1];
  const float* U10 = (const float*)d_in[2];
  const float* U11 = (const float*)d_in[3];
  const float* U20 = (const float*)d_in[4];
  const float* U21 = (const float*)d_in[5];
  const float* U22 = (const float*)d_in[6];
  const float* C    = (const float*)d_in[7];
  const float* beta = (const float*)d_in[8];
  float* out = (float*)d_out;

  (void)hipFuncSetAttribute(reinterpret_cast<const void*>(poly_reg_fused),
                            hipFuncAttributeMaxDynamicSharedMemorySize,
                            SMEM_BYTES);

  dim3 grid(B_ROWS / BLK_ROWS);          // 256 blocks
  dim3 block(256);                       // 8 waves (wave32)
  poly_reg_fused<<<grid, block, SMEM_BYTES, stream>>>(
      X, U00, U10, U11, U20, U21, U22, C, beta, out);
}
